// GraphNetBlock_32487132627389
// MI455X (gfx1250) — matrix operand order repaired and checked by reading, not compiled, not run on hardware
//
#include <hip/hip_runtime.h>
#include <hip/hip_bf16.h>

typedef _Float16 half4_t __attribute__((ext_vector_type(4)));
typedef _Float16 half8_t __attribute__((ext_vector_type(8)));
typedef _Float16 v16h    __attribute__((ext_vector_type(16)));
typedef float    v8f     __attribute__((ext_vector_type(8)));

#define LN_EPS 1e-5f

static __device__ __forceinline__ v8f wmma16(v16h a, v16h b, v8f c) {
    // D = A(16x32) x B(32x16) + C, f32 accumulate
    return __builtin_amdgcn_wmma_f32_16x16x32_f16(
        /*neg_a=*/false, a, /*neg_b=*/false, b,
        /*c_mod=*/(short)0, c, /*reuse_a=*/false, /*reuse_b=*/false);
}

// One MLP layer over a 64-row tile held in LDS (f16, row stride K),
// output 64x128. 128 threads = 4 waves; wave w computes N-tiles {2w, 2w+1}.
// Weights are pre-packed into per-lane B-fragment layout:
//   pw[((kt*8 + nt)*32 + lane)*16 + s] = f16(W[kt*32 + (lane>>4)*16 + s][nt*16 + (lane&15)])
template<int K>
static __device__ __forceinline__ void mlp_layer(
    const _Float16* __restrict__ in,       // LDS [64, K]
    const _Float16* __restrict__ pw,       // global, packed f16 weights
    const float*    __restrict__ bias,     // [128]
    bool relu,
    _Float16* __restrict__ out16,          // LDS [64,128] or nullptr
    float*    __restrict__ out32)          // LDS [64,128] or nullptr
{
    const int tid  = threadIdx.x;
    const int lane = tid & 31;
    const int wave = tid >> 5;
    const int half = lane >> 4;     // K-half selector (this code's contiguous k order, as packed B)
    const int nl   = lane & 15;

    v8f acc[4][2];
    #pragma unroll
    for (int mt = 0; mt < 4; ++mt)
        #pragma unroll
        for (int j = 0; j < 2; ++j)
            acc[mt][j] = (v8f){0.f,0.f,0.f,0.f,0.f,0.f,0.f,0.f};

    for (int kt = 0; kt < K / 32; ++kt) {
        // A fragments: lane holds row M=lane&15; slots 0..15 -> K = k0+half*16..
        // (contiguous order, matching packed B)
        union { v16h v; half8_t h[2]; } afrag[4];
        #pragma unroll
        for (int mt = 0; mt < 4; ++mt) {
            const _Float16* rp = in + (mt * 16 + nl) * K + kt * 32;
            afrag[mt].h[0] = *(const half8_t*)(rp + half * 16);
            afrag[mt].h[1] = *(const half8_t*)(rp + half * 16 + 8);
        }
        #pragma unroll
        for (int j = 0; j < 2; ++j) {
            const int nt = wave * 2 + j;
            v16h b = *(const v16h*)(pw + ((size_t)((kt * 8 + nt) * 32 + lane)) * 16);
            #pragma unroll
            for (int mt = 0; mt < 4; ++mt)
                acc[mt][j] = wmma16(afrag[mt].v, b, acc[mt][j]);
        }
    }

    // Epilogue: C/D layout -> VGPR r, lanes 0-15: M=r, lanes 16-31: M=8+r; N=lane&15
    #pragma unroll
    for (int j = 0; j < 2; ++j) {
        const int nt  = wave * 2 + j;
        const int col = nt * 16 + nl;
        const float bv = bias[col];
        #pragma unroll
        for (int mt = 0; mt < 4; ++mt) {
            #pragma unroll
            for (int r = 0; r < 8; ++r) {
                float v = acc[mt][j][r] + bv;
                if (relu) v = fmaxf(v, 0.f);
                const int row = mt * 16 + half * 8 + r;
                if (out16) out16[row * 128 + col] = (_Float16)v;
                if (out32) out32[row * 128 + col] = v;
            }
        }
    }
}

// -------- weight packing: W[Kdim,128] fp32 -> f16 B-fragment layout --------
__global__ void pack_weights_kernel(const float* __restrict__ W,
                                    _Float16* __restrict__ out, int nkt) {
    int i = blockIdx.x * blockDim.x + threadIdx.x;
    int total = nkt * 8 * 32 * 16;
    if (i >= total) return;
    int s    = i & 15;
    int lane = (i >> 4) & 31;
    int nt   = (i >> 9) & 7;
    int kt   = i >> 12;
    int k = kt * 32 + (lane >> 4) * 16 + s;
    int n = nt * 16 + (lane & 15);
    out[i] = (_Float16)W[k * 128 + n];
}

__global__ void zero_kernel(float4* __restrict__ p, long long n4) {
    long long i = (long long)blockIdx.x * blockDim.x + threadIdx.x;
    if (i < n4) p[i] = make_float4(0.f, 0.f, 0.f, 0.f);
}

// ------------------------------ edge kernel -------------------------------
__global__ __launch_bounds__(128) void edge_mlp_kernel(
    const int*   __restrict__ senders,   const int* __restrict__ receivers,
    const float* __restrict__ nodef,     const float* __restrict__ edgef,
    const _Float16* __restrict__ pW0, const _Float16* __restrict__ pW1,
    const _Float16* __restrict__ pW2,
    const float* __restrict__ b0, const float* __restrict__ b1,
    const float* __restrict__ b2,
    const float* __restrict__ gamma, const float* __restrict__ beta,
    float* __restrict__ acc, float* __restrict__ out_edge, int E)
{
    __shared__ __align__(16) char s_raw[64 * 384 * 2];     // edge_in f16 / h3 f32
    __shared__ __align__(16) _Float16 s_h1[64 * 128];
    __shared__ __align__(16) _Float16 s_h2[64 * 128];
    __shared__ int   s_send[64];
    __shared__ int   s_recv[64];
    __shared__ float s_mu[64], s_rs[64];

    _Float16* s_in = (_Float16*)s_raw;
    float*    h3   = (float*)s_raw;

    const int tid = threadIdx.x;
    const int e0  = blockIdx.x * 64;

    if (tid < 64) {
        int ge = e0 + tid;
        s_send[tid] = (ge < E) ? senders[ge]   : 0;
        s_recv[tid] = (ge < E) ? receivers[ge] : -1;
    }
    __syncthreads();

    // Stage edge_in = [sender_feat | receiver_feat | edge_feat] as f16 in LDS
    for (int i = tid; i < 64 * 96; i += 128) {   // 96 float4 per row of 384
        int e  = i / 96, c4 = i - e * 96;
        int seg = c4 >> 5, cc = (c4 & 31) << 2;
        int ge = e0 + e;
        float4 v = make_float4(0.f, 0.f, 0.f, 0.f);
        if (ge < E) {
            const float* src;
            if (seg == 0)      src = nodef + (size_t)s_send[e] * 128 + cc;
            else if (seg == 1) src = nodef + (size_t)s_recv[e] * 128 + cc;
            else               src = edgef + (size_t)ge * 128 + cc;
            v = *(const float4*)src;
        }
        half4_t h;
        h[0] = (_Float16)v.x; h[1] = (_Float16)v.y;
        h[2] = (_Float16)v.z; h[3] = (_Float16)v.w;
        *(half4_t*)(&s_in[e * 384 + c4 * 4]) = h;
    }
    __syncthreads();

    mlp_layer<384>(s_in, pW0, b0, true,  s_h1, nullptr);
    __syncthreads();
    mlp_layer<128>(s_h1, pW1, b1, true,  s_h2, nullptr);
    __syncthreads();
    mlp_layer<128>(s_h2, pW2, b2, false, nullptr, h3);   // h3 aliases s_in (dead)
    __syncthreads();

    // LayerNorm statistics per edge row
    if (tid < 64) {
        const float* r = h3 + tid * 128;
        float s = 0.f;
        for (int c = 0; c < 128; ++c) s += r[c];
        float mu = s * (1.f / 128.f);
        float vs = 0.f;
        for (int c = 0; c < 128; ++c) { float d = r[c] - mu; vs += d * d; }
        s_mu[tid] = mu;
        s_rs[tid] = rsqrtf(vs * (1.f / 128.f) + LN_EPS);
    }
    __syncthreads();

    // normalize, residual store, scatter-add to receiver nodes
    for (int i = tid; i < 64 * 128; i += 128) {
        int e = i >> 7, c = i & 127;
        int ge = e0 + e;
        if (ge >= E) continue;
        float ne = (h3[i] - s_mu[e]) * s_rs[e] * gamma[c] + beta[c];
        out_edge[(size_t)ge * 128 + c] = ne + edgef[(size_t)ge * 128 + c];
        int rcv = s_recv[e];
        if (rcv >= 0)
            unsafeAtomicAdd(&acc[(size_t)rcv * 128 + c], ne);
    }
}

// ------------------------------ node kernel -------------------------------
__global__ __launch_bounds__(128) void node_mlp_kernel(
    const float* __restrict__ nodef, const float* __restrict__ acc,
    const _Float16* __restrict__ pW0, const _Float16* __restrict__ pW1,
    const _Float16* __restrict__ pW2,
    const float* __restrict__ b0, const float* __restrict__ b1,
    const float* __restrict__ b2,
    const float* __restrict__ gamma, const float* __restrict__ beta,
    float* __restrict__ out_node, int N)
{
    __shared__ __align__(16) char s_raw[64 * 256 * 2];   // node_in f16 / h3 f32
    __shared__ __align__(16) _Float16 s_h1[64 * 128];
    __shared__ __align__(16) _Float16 s_h2[64 * 128];
    __shared__ float s_mu[64], s_rs[64];

    _Float16* s_in = (_Float16*)s_raw;
    float*    h3   = (float*)s_raw;

    const int tid = threadIdx.x;
    const int n0  = blockIdx.x * 64;

    // Stage node_in = [node_feat | acc] as f16
    for (int i = tid; i < 64 * 64; i += 128) {   // 64 float4 per row of 256
        int r = i >> 6, c4 = i & 63;
        int seg = c4 >> 5, cc = (c4 & 31) << 2;
        int g = n0 + r;
        float4 v = make_float4(0.f, 0.f, 0.f, 0.f);
        if (g < N) {
            const float* src = (seg == 0) ? (nodef + (size_t)g * 128 + cc)
                                          : (acc   + (size_t)g * 128 + cc);
            v = *(const float4*)src;
        }
        half4_t h;
        h[0] = (_Float16)v.x; h[1] = (_Float16)v.y;
        h[2] = (_Float16)v.z; h[3] = (_Float16)v.w;
        *(half4_t*)(&s_in[r * 256 + c4 * 4]) = h;
    }
    __syncthreads();

    mlp_layer<256>(s_in, pW0, b0, true,  s_h1, nullptr);
    __syncthreads();
    mlp_layer<128>(s_h1, pW1, b1, true,  s_h2, nullptr);
    __syncthreads();
    mlp_layer<128>(s_h2, pW2, b2, false, nullptr, h3);   // aliases s_in (dead)
    __syncthreads();

    if (tid < 64) {
        const float* r = h3 + tid * 128;
        float s = 0.f;
        for (int c = 0; c < 128; ++c) s += r[c];
        float mu = s * (1.f / 128.f);
        float vs = 0.f;
        for (int c = 0; c < 128; ++c) { float d = r[c] - mu; vs += d * d; }
        s_mu[tid] = mu;
        s_rs[tid] = rsqrtf(vs * (1.f / 128.f) + LN_EPS);
    }
    __syncthreads();

    for (int i = tid; i < 64 * 128; i += 128) {
        int r = i >> 7, c = i & 127;
        int g = n0 + r;
        if (g >= N) continue;
        float nn = (h3[i] - s_mu[r]) * s_rs[r] * gamma[c] + beta[c];
        out_node[(size_t)g * 128 + c] = nn + nodef[(size_t)g * 128 + c];
    }
}

// ------------------------------- launcher ---------------------------------
extern "C" void kernel_launch(void* const* d_in, const int* in_sizes, int n_in,
                              void* d_out, int out_size, void* d_ws, size_t ws_size,
                              hipStream_t stream) {
    const int*   senders   = (const int*)d_in[0];
    const int*   receivers = (const int*)d_in[1];
    const float* nodef     = (const float*)d_in[2];
    const float* edgef     = (const float*)d_in[3];
    const float* eW0 = (const float*)d_in[4];
    const float* eb0 = (const float*)d_in[5];
    const float* eW1 = (const float*)d_in[6];
    const float* eb1 = (const float*)d_in[7];
    const float* eW2 = (const float*)d_in[8];
    const float* eb2 = (const float*)d_in[9];
    const float* eg    = (const float*)d_in[10];
    const float* ebeta = (const float*)d_in[11];
    const float* nW0 = (const float*)d_in[12];
    const float* nb0 = (const float*)d_in[13];
    const float* nW1 = (const float*)d_in[14];
    const float* nb1 = (const float*)d_in[15];
    const float* nW2 = (const float*)d_in[16];
    const float* nb2 = (const float*)d_in[17];
    const float* ng    = (const float*)d_in[18];
    const float* nbeta = (const float*)d_in[19];

    const int E = in_sizes[0];            // B=1
    const int N = in_sizes[2] / 128;

    // workspace layout: [acc: N*128 f32][packed f16 weights]
    float* acc = (float*)d_ws;
    size_t accBytes = ((size_t)N * 128 * sizeof(float) + 255) & ~(size_t)255;
    _Float16* pw   = (_Float16*)((char*)d_ws + accBytes);
    _Float16* peW0 = pw;                    // 12 ktiles
    _Float16* peW1 = peW0 + (size_t)12 * 4096;
    _Float16* peW2 = peW1 + (size_t)4  * 4096;
    _Float16* pnW0 = peW2 + (size_t)4  * 4096;  // 8 ktiles
    _Float16* pnW1 = pnW0 + (size_t)8  * 4096;
    _Float16* pnW2 = pnW1 + (size_t)4  * 4096;

    pack_weights_kernel<<<(12 * 4096 + 255) / 256, 256, 0, stream>>>(eW0, peW0, 12);
    pack_weights_kernel<<<( 4 * 4096 + 255) / 256, 256, 0, stream>>>(eW1, peW1, 4);
    pack_weights_kernel<<<( 4 * 4096 + 255) / 256, 256, 0, stream>>>(eW2, peW2, 4);
    pack_weights_kernel<<<( 8 * 4096 + 255) / 256, 256, 0, stream>>>(nW0, pnW0, 8);
    pack_weights_kernel<<<( 4 * 4096 + 255) / 256, 256, 0, stream>>>(nW1, pnW1, 4);
    pack_weights_kernel<<<( 4 * 4096 + 255) / 256, 256, 0, stream>>>(nW2, pnW2, 4);

    long long n4 = (long long)N * 32;     // N*128/4 float4s
    zero_kernel<<<(unsigned)((n4 + 255) / 256), 256, 0, stream>>>((float4*)acc, n4);

    float* out_node = (float*)d_out;
    float* out_edge = out_node + (size_t)N * 128;

    edge_mlp_kernel<<<(E + 63) / 64, 128, 0, stream>>>(
        senders, receivers, nodef, edgef, peW0, peW1, peW2,
        eb0, eb1, eb2, eg, ebeta, acc, out_edge, E);

    node_mlp_kernel<<<(N + 63) / 64, 128, 0, stream>>>(
        nodef, acc, pnW0, pnW1, pnW2, nb0, nb1, nb2, ng, nbeta, out_node, N);
}
